// TSCluster_65996467471169
// MI455X (gfx1250) — compile-verified, matrix operand order checked by reading, not run on hardware
//
#include <hip/hip_runtime.h>
#include <hip/hip_bf16.h>

typedef _Float16 h8   __attribute__((ext_vector_type(8)));
typedef _Float16 v16h __attribute__((ext_vector_type(16)));
typedef float    v8f  __attribute__((ext_vector_type(8)));

#define T_TOKENS 131072   // N*P = 2048*64
#define N_SER    2048
#define N_LAYERS 3
#define LN_EPSF  1e-5f
#define EPSF     1e-10f

union FragU { v16h v; struct { h8 lo; h8 hi; } p; };

__device__ __forceinline__ v8f v8f_zero() {
  v8f a;
#pragma unroll
  for (int i = 0; i < 8; ++i) a[i] = 0.0f;
  return a;
}
__device__ __forceinline__ h8 h8_zero() {
  h8 a;
#pragma unroll
  for (int i = 0; i < 8; ++i) a[i] = (_Float16)0.0f;
  return a;
}

// ---- CDNA5 async global->LDS copy (ASYNCcnt path, CDNA5 ISA) ----
// vdst = LDS byte address (low 32 bits of generic shared pointer), vaddr = 64b global.
__device__ __forceinline__ void async_load_b128(const _Float16* gptr, _Float16* lptr) {
  const unsigned lds = (unsigned)(unsigned long long)lptr;
  asm volatile("global_load_async_to_lds_b128 %0, %1, off"
               :: "v"(lds), "v"(gptr) : "memory");
}
__device__ __forceinline__ void wait_async0() {
  asm volatile("s_wait_asynccnt 0x0" ::: "memory");
}

// Build a 16x32 (f16) WMMA A/B fragment from an LDS tile, per CDNA5 ISA layout:
// lane L: row = base + (L&15); halves [8*(L>>4) .. +7] and [16 + 8*(L>>4) .. +7].
template<int LDH>
__device__ __forceinline__ v16h load_frag(const _Float16* tile, int row_base, int k0, int lane) {
  const int r    = row_base + (lane & 15);
  const int hofs = (lane >> 4) << 3;
  const _Float16* p = tile + r * LDH + k0 + hofs;
  FragU u;
  u.p.lo = *(const h8*)(p);
  u.p.hi = *(const h8*)(p + 16);
  return u.v;
}

__device__ __forceinline__ v8f wmma16(v16h a, v16h b, v8f c) {
  return __builtin_amdgcn_wmma_f32_16x16x32_f16(false, a, false, b, (short)0, c, false, false);
}

// ---------------- generic WMMA GEMM: C[M,N] = A[M,K] @ B[N,K]^T + bias ----------------
// 128x128 block tile, BK=32, 8 waves; each wave owns 32x64 (2x4 WMMA tiles).
// Tiles staged via double-buffered global_load_async_to_lds_b128.
// MODE 0: store f32, MODE 1: store f16, MODE 2: relu + store f16
template<int MODE>
__global__ __launch_bounds__(256) void wmma_gemm_kernel(
    const _Float16* __restrict__ A, const _Float16* __restrict__ B,
    const float* __restrict__ bias, void* __restrict__ Cout,
    int M, int N, int K)
{
  constexpr int LDS_P = 48;  // padded row (halves); 96B rows keep b128 16B-aligned
  __shared__ __align__(16) _Float16 sA[2][128 * LDS_P];
  __shared__ __align__(16) _Float16 sB[2][128 * LDS_P];

  const int tid  = threadIdx.x;
  const int lane = tid & 31;
  const int wave = tid >> 5;               // 8 waves
  const int m0 = blockIdx.y * 128;
  const int n0 = blockIdx.x * 128;
  const int wm = (wave & 3) << 5;          // wave M offset: 0/32/64/96
  const int wn = (wave >> 2) << 6;         // wave N offset: 0/64

  v8f acc[2][4];
#pragma unroll
  for (int mi = 0; mi < 2; ++mi)
#pragma unroll
    for (int ni = 0; ni < 4; ++ni) acc[mi][ni] = v8f_zero();

  // tile stage-in: each thread owns one 16-half row segment (2 x b128)
  const int lr = tid >> 1;                 // 0..127 (tile row)
  const int lc = (tid & 1) << 4;           // 0 or 16 (half offset)  -> covers 32 halves/row
  const size_t arow = (size_t)(m0 + lr) * K;
  const bool   bok  = (n0 + lr) < N;
  const size_t brow = (size_t)(n0 + lr) * K;

  auto load_tile = [&](int k0, int buf) {
    _Float16* dA = &sA[buf][lr * LDS_P + lc];
    async_load_b128(A + arow + k0 + lc,     dA);
    async_load_b128(A + arow + k0 + lc + 8, dA + 8);
    _Float16* dB = &sB[buf][lr * LDS_P + lc];
    if (bok) {
      async_load_b128(B + brow + k0 + lc,     dB);
      async_load_b128(B + brow + k0 + lc + 8, dB + 8);
    } else {
      *(h8*)dB = h8_zero();
      *(h8*)(dB + 8) = h8_zero();
    }
  };

  load_tile(0, 0);
  wait_async0();
  __syncthreads();

  int buf = 0;
  for (int k0 = 0; k0 < K; k0 += 32) {
    if (k0 + 32 < K) load_tile(k0 + 32, buf ^ 1);   // prefetch next tile (async)

    v16h a0 = load_frag<LDS_P>(sA[buf], wm,      0, lane);
    v16h a1 = load_frag<LDS_P>(sA[buf], wm + 16, 0, lane);
    v16h b0 = load_frag<LDS_P>(sB[buf], wn,      0, lane);
    v16h b1 = load_frag<LDS_P>(sB[buf], wn + 16, 0, lane);
    v16h b2 = load_frag<LDS_P>(sB[buf], wn + 32, 0, lane);
    v16h b3 = load_frag<LDS_P>(sB[buf], wn + 48, 0, lane);

    acc[0][0] = wmma16(a0, b0, acc[0][0]);
    acc[0][1] = wmma16(a0, b1, acc[0][1]);
    acc[0][2] = wmma16(a0, b2, acc[0][2]);
    acc[0][3] = wmma16(a0, b3, acc[0][3]);
    acc[1][0] = wmma16(a1, b0, acc[1][0]);
    acc[1][1] = wmma16(a1, b1, acc[1][1]);
    acc[1][2] = wmma16(a1, b2, acc[1][2]);
    acc[1][3] = wmma16(a1, b3, acc[1][3]);

    wait_async0();       // next buffer resident
    __syncthreads();     // everyone done reading current buffer
    buf ^= 1;
  }

  const int half = lane >> 4;
#pragma unroll
  for (int ni = 0; ni < 4; ++ni) {
    const int c = n0 + wn + ni * 16 + (lane & 15);
    const float bb = (bias && c < N) ? bias[c] : 0.0f;
#pragma unroll
    for (int mi = 0; mi < 2; ++mi) {
#pragma unroll
      for (int r = 0; r < 8; ++r) {
        const int cm = m0 + wm + mi * 16 + r + (half << 3);
        float v = acc[mi][ni][r] + bb;
        if (MODE == 2) v = fmaxf(v, 0.0f);
        if (c < N) {
          if (MODE == 0) ((float*)Cout)[(size_t)cm * N + c] = v;
          else           ((_Float16*)Cout)[(size_t)cm * N + c] = (_Float16)v;
        }
      }
    }
  }
}

// ---------------- attention: one wave per (series n, head h); P=64, dh=64 ----------------
__global__ __launch_bounds__(32) void attention_kernel(
    const _Float16* __restrict__ qkv, _Float16* __restrict__ out)
{
  constexpr int LDH = 72;  // 144B rows (16B multiple)
  __shared__ __align__(16) _Float16 sQ [64 * LDH];  // also reused for softmax probs
  __shared__ __align__(16) _Float16 sK [64 * LDH];
  __shared__ __align__(16) _Float16 sVT[64 * LDH];  // V^T: [dh][p]
  __shared__ float sS[64 * 68];

  const int lane = threadIdx.x;
  const int n = blockIdx.x >> 3;
  const int h = blockIdx.x & 7;
  const size_t t0 = (size_t)n * 64;
  const int qc = h * 64, kc = 512 + h * 64, vc = 1024 + h * 64;

#pragma unroll
  for (int rr = 0; rr < 2; ++rr) {
    const int row = lane * 2 + rr;
    const _Float16* gq = qkv + (t0 + row) * 1536 + qc;
    const _Float16* gk = qkv + (t0 + row) * 1536 + kc;
#pragma unroll
    for (int j = 0; j < 8; ++j) {
      *(h8*)(sQ + row * LDH + j * 8) = *(const h8*)(gq + j * 8);
      *(h8*)(sK + row * LDH + j * 8) = *(const h8*)(gk + j * 8);
    }
  }
#pragma unroll
  for (int rr = 0; rr < 2; ++rr) {          // V loaded transposed
    const int d = lane * 2 + rr;
    for (int p = 0; p < 64; ++p)
      sVT[d * LDH + p] = qkv[(t0 + p) * 1536 + vc + d];
  }
  __syncthreads();

  const int half = lane >> 4;
  const int cn   = lane & 15;

  // S = (Q @ K^T) / sqrt(64)
  for (int mt = 0; mt < 4; ++mt)
    for (int nt = 0; nt < 4; ++nt) {
      v8f acc = v8f_zero();
#pragma unroll
      for (int kk = 0; kk < 64; kk += 32) {
        v16h af = load_frag<LDH>(sQ, mt * 16, kk, lane);
        v16h bf = load_frag<LDH>(sK, nt * 16, kk, lane);
        acc = wmma16(af, bf, acc);
      }
#pragma unroll
      for (int r = 0; r < 8; ++r)
        sS[(mt * 16 + r + (half << 3)) * 68 + nt * 16 + cn] = acc[r] * 0.125f;
    }
  __syncthreads();

  // row softmax -> f16 probs into sQ (Q no longer needed)
#pragma unroll
  for (int rr = 0; rr < 2; ++rr) {
    const int row = lane + rr * 32;
    float m = -3.0e38f;
    for (int j = 0; j < 64; ++j) m = fmaxf(m, sS[row * 68 + j]);
    float s = 0.0f;
    for (int j = 0; j < 64; ++j) { float e = __expf(sS[row * 68 + j] - m); sS[row * 68 + j] = e; s += e; }
    const float inv = 1.0f / s;
    for (int j = 0; j < 64; ++j) sQ[row * LDH + j] = (_Float16)(sS[row * 68 + j] * inv);
  }
  __syncthreads();

  // O = P @ V^T  (both operands K-major over keys)
  for (int mt = 0; mt < 4; ++mt)
    for (int nt = 0; nt < 4; ++nt) {
      v8f acc = v8f_zero();
#pragma unroll
      for (int kk = 0; kk < 64; kk += 32) {
        v16h af = load_frag<LDH>(sQ,  mt * 16, kk, lane);
        v16h bf = load_frag<LDH>(sVT, nt * 16, kk, lane);
        acc = wmma16(af, bf, acc);
      }
#pragma unroll
      for (int r = 0; r < 8; ++r) {
        const int row = mt * 16 + r + (half << 3);
        const int col = h * 64 + nt * 16 + cn;
        out[(t0 + row) * 512 + col] = (_Float16)acc[r];
      }
    }
}

// ---------------- LayerNorm(x + o): one block per token row ----------------
__global__ __launch_bounds__(256) void layernorm_kernel(
    const float* __restrict__ x, const float* __restrict__ o,
    const float* __restrict__ w, const float* __restrict__ b,
    float* __restrict__ xout, _Float16* __restrict__ hout)
{
  __shared__ float red[256];
  const size_t row = blockIdx.x;
  const int tid = threadIdx.x;
  const float y0 = x[row * 512 + tid]       + o[row * 512 + tid];
  const float y1 = x[row * 512 + tid + 256] + o[row * 512 + tid + 256];
  red[tid] = y0 + y1;
  __syncthreads();
  for (int s = 128; s > 0; s >>= 1) { if (tid < s) red[tid] += red[tid + s]; __syncthreads(); }
  const float mean = red[0] * (1.0f / 512.0f);
  __syncthreads();
  red[tid] = (y0 - mean) * (y0 - mean) + (y1 - mean) * (y1 - mean);
  __syncthreads();
  for (int s = 128; s > 0; s >>= 1) { if (tid < s) red[tid] += red[tid + s]; __syncthreads(); }
  const float rstd = rsqrtf(red[0] * (1.0f / 512.0f) + LN_EPSF);
  const float r0 = (y0 - mean) * rstd * w[tid]       + b[tid];
  const float r1 = (y1 - mean) * rstd * w[tid + 256] + b[tid + 256];
  xout[row * 512 + tid]       = r0;
  xout[row * 512 + tid + 256] = r1;
  hout[row * 512 + tid]       = (_Float16)r0;
  hout[row * 512 + tid + 256] = (_Float16)r1;
}

__global__ void cast_f32_to_f16_kernel(const float* __restrict__ in,
                                       _Float16* __restrict__ out, size_t n)
{
  size_t i = (size_t)blockIdx.x * blockDim.x + threadIdx.x;
  const size_t stride = (size_t)gridDim.x * blockDim.x;
  for (; i < n; i += stride) out[i] = (_Float16)in[i];
}

// mu = tanh(mus) -> f16, musq[c] = sum(mu^2). One wave per cluster.
__global__ __launch_bounds__(256) void mu_kernel(const float* __restrict__ mus,
    _Float16* __restrict__ mu_h, float* __restrict__ musq)
{
  const int wave = threadIdx.x >> 5, lane = threadIdx.x & 31;
  const int c = blockIdx.x * 8 + wave;
  float acc = 0.0f;
  for (int i = lane; i < 512; i += 32) {
    const float m = tanhf(mus[c * 512 + i]);
    mu_h[c * 512 + i] = (_Float16)m;
    acc += m * m;
  }
  for (int o = 16; o > 0; o >>= 1) acc += __shfl_xor(acc, o);
  if (lane == 0) musq[c] = acc;
}

// per token: logits over 32 clusters (lane == cluster), write log(softmax + eps)
__global__ __launch_bounds__(256) void cluster_logits_kernel(
    const float* __restrict__ x, const float* __restrict__ xmu,
    const float* __restrict__ musq, float* __restrict__ log_za)
{
  const int wave = threadIdx.x >> 5, lane = threadIdx.x & 31;
  const size_t t = (size_t)blockIdx.x * 8 + wave;
  float sq = 0.0f;
  for (int i = lane; i < 512; i += 32) { const float v = x[t * 512 + i]; sq += v * v; }
  for (int o = 16; o > 0; o >>= 1) sq += __shfl_xor(sq, o);
  const float dist = sq + musq[lane] - 2.0f * xmu[t * 32 + lane];
  const float z = __expf(-dist * 0.1f);           // TEMP = 10
  float s = z;
  for (int o = 16; o > 0; o >>= 1) s += __shfl_xor(s, o);
  log_za[t * 32 + lane] = __logf(z / s + EPSF);
}

__global__ void tail_kernel(const float* __restrict__ pi, const float* __restrict__ A,
                            float* __restrict__ out_pi, float* __restrict__ out_A,
                            float* __restrict__ loss)
{
  const int i = blockIdx.x * blockDim.x + threadIdx.x;
  if (i < 32)   out_pi[i] = pi[i];
  if (i < 1024) out_A[i]  = A[i];
  if (i == 0)   *loss = 0.0f;
}

// HMM recursion: one wave per series; lane == cluster (C == 32 == wave32)
__global__ __launch_bounds__(256) void hmm_kernel(
    const float* __restrict__ log_za, const float* __restrict__ pi,
    const float* __restrict__ A, float* __restrict__ cp, float* __restrict__ loss)
{
  __shared__ float sA[1024];
  const int tid = threadIdx.x;
  for (int i = tid; i < 1024; i += 256) sA[i] = A[i];
  __syncthreads();
  const int wave = tid >> 5, lane = tid & 31;
  const int n = blockIdx.x * 8 + wave;
  const size_t base = (size_t)n * 64 * 32;

  float lp = __logf(pi[lane] + EPSF) + log_za[base + lane];
  float m = lp;
  for (int o = 16; o > 0; o >>= 1) m = fmaxf(m, __shfl_xor(m, o));
  float e = __expf(lp - m);
  float s = e;
  for (int o = 16; o > 0; o >>= 1) s += __shfl_xor(s, o);
  float c = e / s;
  cp[base + lane] = c;
  float ent = c * __logf(c + EPSF);

  for (int p = 1; p < 64; ++p) {
    float t = 0.0f;
#pragma unroll
    for (int j = 0; j < 32; ++j)
      t += __shfl(c, j) * sA[j * 32 + lane];     // (c @ A)[lane]
    lp = __logf(t + EPSF) + log_za[base + (size_t)p * 32 + lane];
    m = lp;
    for (int o = 16; o > 0; o >>= 1) m = fmaxf(m, __shfl_xor(m, o));
    e = __expf(lp - m);
    s = e;
    for (int o = 16; o > 0; o >>= 1) s += __shfl_xor(s, o);
    c = e / s;
    cp[base + (size_t)p * 32 + lane] = c;
    ent += c * __logf(c + EPSF);
  }
  for (int o = 16; o > 0; o >>= 1) ent += __shfl_xor(ent, o);
  if (lane == 0) atomicAdd(loss, -ent * (1.0f / (2048.0f * 64.0f)));
}

// ------------------------------------------------------------------------------------
extern "C" void kernel_launch(void* const* d_in, const int* in_sizes, int n_in,
                              void* d_out, int out_size, void* d_ws, size_t ws_size,
                              hipStream_t stream)
{
  const float* in_x  = (const float*)d_in[0];
  const float* pi    = (const float*)d_in[1];
  const float* Amat  = (const float*)d_in[2];
  const float* qkv_w = (const float*)d_in[3];
  const float* qkv_b = (const float*)d_in[4];
  const float* out_w = (const float*)d_in[5];
  const float* out_b = (const float*)d_in[6];
  const float* ln1_w = (const float*)d_in[7];
  const float* ln1_b = (const float*)d_in[8];
  const float* ff1_w = (const float*)d_in[9];
  const float* ff1_b = (const float*)d_in[10];
  const float* ff2_w = (const float*)d_in[11];
  const float* ff2_b = (const float*)d_in[12];
  const float* ln2_w = (const float*)d_in[13];
  const float* ln2_b = (const float*)d_in[14];
  const float* mus   = (const float*)d_in[15];

  const size_t T = T_TOKENS;
  char* ws = (char*)d_ws;
  size_t off = 0;
  auto carve = [&](size_t bytes) -> void* {
    void* p = ws + off;
    off += (bytes + 255) & ~(size_t)255;
    return p;
  };

  float*     x_f32  = (float*)    carve(T * 512 * 4);
  _Float16*  x_f16  = (_Float16*) carve(T * 512 * 2);
  _Float16*  qkv_h  = (_Float16*) carve(T * 1536 * 2);
  _Float16*  o_h    = (_Float16*) carve(T * 512 * 2);
  float*     o_f32  = (float*)    carve(T * 512 * 4);
  _Float16*  ffh    = (_Float16*) carve(T * 2048 * 2);
  float*     xmu    = (float*)    carve(T * 32 * 4);
  float*     logza  = (float*)    carve(T * 32 * 4);
  _Float16*  mu_h   = (_Float16*) carve(32 * 512 * 2);
  float*     musq   = (float*)    carve(32 * 4);
  _Float16*  qkvw_h = (_Float16*) carve((size_t)3 * 1536 * 512 * 2);
  _Float16*  outw_h = (_Float16*) carve((size_t)3 * 512 * 512 * 2);
  _Float16*  ff1w_h = (_Float16*) carve((size_t)3 * 2048 * 512 * 2);
  _Float16*  ff2w_h = (_Float16*) carve((size_t)3 * 512 * 2048 * 2);

  // x = input (f32 residual stream) + f16 mirror for GEMM A
  hipMemcpyAsync(x_f32, in_x, T * 512 * 4, hipMemcpyDeviceToDevice, stream);
  cast_f32_to_f16_kernel<<<4096, 256, 0, stream>>>(in_x, x_f16, T * 512);
  // f16 weights (per launch; 19 MB total)
  cast_f32_to_f16_kernel<<<2048, 256, 0, stream>>>(qkv_w, qkvw_h, (size_t)3 * 1536 * 512);
  cast_f32_to_f16_kernel<<<1024, 256, 0, stream>>>(out_w, outw_h, (size_t)3 * 512 * 512);
  cast_f32_to_f16_kernel<<<2048, 256, 0, stream>>>(ff1_w, ff1w_h, (size_t)3 * 2048 * 512);
  cast_f32_to_f16_kernel<<<2048, 256, 0, stream>>>(ff2_w, ff2w_h, (size_t)3 * 512 * 2048);

  const dim3 blk(256);
  for (int l = 0; l < N_LAYERS; ++l) {
    wmma_gemm_kernel<1><<<dim3(1536 / 128, T / 128), blk, 0, stream>>>(
        x_f16, qkvw_h + (size_t)l * 1536 * 512, qkv_b + l * 1536, qkv_h, (int)T, 1536, 512);
    attention_kernel<<<N_SER * 8, 32, 0, stream>>>(qkv_h, o_h);
    wmma_gemm_kernel<0><<<dim3(512 / 128, T / 128), blk, 0, stream>>>(
        o_h, outw_h + (size_t)l * 512 * 512, out_b + l * 512, o_f32, (int)T, 512, 512);
    layernorm_kernel<<<T, 256, 0, stream>>>(x_f32, o_f32, ln1_w + l * 512, ln1_b + l * 512, x_f32, x_f16);
    wmma_gemm_kernel<2><<<dim3(2048 / 128, T / 128), blk, 0, stream>>>(
        x_f16, ff1w_h + (size_t)l * 2048 * 512, ff1_b + l * 2048, ffh, (int)T, 2048, 512);
    wmma_gemm_kernel<0><<<dim3(512 / 128, T / 128), blk, 0, stream>>>(
        ffh, ff2w_h + (size_t)l * 512 * 2048, ff2_b + l * 512, o_f32, (int)T, 512, 2048);
    layernorm_kernel<<<T, 256, 0, stream>>>(x_f32, o_f32, ln2_w + l * 512, ln2_b + l * 512, x_f32, x_f16);
  }

  // output layout: cluster_probs | emb_ts | pi | A | loss
  float* out_cp   = (float*)d_out;
  float* out_emb  = out_cp + (size_t)2048 * 64 * 32;
  float* out_pi   = out_emb + (size_t)T * 512;
  float* out_A    = out_pi + 32;
  float* out_loss = out_A + 1024;

  hipMemcpyAsync(out_emb, x_f32, T * 512 * 4, hipMemcpyDeviceToDevice, stream);

  mu_kernel<<<4, 256, 0, stream>>>(mus, mu_h, musq);
  wmma_gemm_kernel<0><<<dim3(1, T / 128), blk, 0, stream>>>(
      x_f16, mu_h, nullptr, xmu, (int)T, 32, 512);
  cluster_logits_kernel<<<T / 8, 256, 0, stream>>>(x_f32, xmu, musq, logza);
  tail_kernel<<<4, 256, 0, stream>>>(pi, Amat, out_pi, out_A, out_loss);
  hmm_kernel<<<N_SER / 8, 256, 0, stream>>>(logza, pi, Amat, out_cp, out_loss);
}
